// WalletGNN_5677946765508
// MI455X (gfx1250) — matrix operand order repaired and checked by reading, not compiled, not run on hardware
//
#include <hip/hip_runtime.h>
#include <hip/hip_bf16.h>
#include <hip/hip_fp16.h>

typedef __attribute__((ext_vector_type(16))) _Float16 v16h;
typedef __attribute__((ext_vector_type(8)))  _Float16 v8h;
typedef __attribute__((ext_vector_type(8)))  float    v8f;
typedef __attribute__((ext_vector_type(4)))  float    f32x4;

#define HIDC 128

// =====================================================================
// WMMA GEMM: C[M,Nc] = act(A[M,K] @ B[K,Nc] + bias)
// Requirements: K % 32 == 0, Nc % 8 == 0.
// Block: 256 threads = 8 waves; block tile 64x64; wave tile 16x32.
// f16 inputs (converted during LDS staging), f32 accumulate via
// v_wmma_f32_16x16x32_f16. All LDS traffic is 16-byte b128 ops.
// OOB rows/cols are clamped to row0/col0 (always in-bounds); the
// resulting garbage only lands in output elements the guarded epilogue
// never stores, so the hot loop is branch-free.
// =====================================================================
__global__ __launch_bounds__(256)
void gemm_wmma(const float* __restrict__ A, const float* __restrict__ B,
               const float* __restrict__ bias, float* __restrict__ C,
               int M, int K, int Nc, int act)
{
    __shared__ _Float16 As[64][40];   // [row][k]  (pad 40 -> 80B row stride)
    __shared__ _Float16 Bt[64][40];   // [col][k]  (transposed B tile)
    const int tid  = threadIdx.x;
    const int lane = tid & 31;
    const int wid  = tid >> 5;
    const int wm   = wid & 3;         // row tile 0..3
    const int wn   = wid >> 2;        // col group 0..1 (32 cols)
    const int row0 = blockIdx.y * 64;
    const int col0 = blockIdx.x * 64;
    const int half = lane >> 4;
    const int r16  = lane & 15;

    // A staging: thread t -> row ar = t>>2, 8-col chunk ac0 = (t&3)*8
    const int ar  = tid >> 2;
    const int ac0 = (tid & 3) * 8;
    const int arow_clamped = (row0 + ar) < M ? (row0 + ar) : 0;
    const float* Ap = A + (size_t)arow_clamped * K + ac0;

    // B staging: thread t -> col bc = t&63, 8-row k chunk bk0 = (t>>6)*8
    const int bc  = tid & 63;
    const int bk0 = (tid >> 6) * 8;
    const int bcol_clamped = (col0 + bc) < Nc ? (col0 + bc) : 0;
    const float* Bp = B + (size_t)bk0 * Nc + bcol_clamped;

    v8f c0 = {};
    v8f c1 = {};

    for (int kk = 0; kk < K; kk += 32) {
        // ---- A tile: 2x b128 load, convert, 1x b128 LDS store ----
        f32x4 a0 = *(const f32x4*)(Ap + 0);
        f32x4 a1 = *(const f32x4*)(Ap + 4);
        v8h ah = { (_Float16)a0[0], (_Float16)a0[1], (_Float16)a0[2], (_Float16)a0[3],
                   (_Float16)a1[0], (_Float16)a1[1], (_Float16)a1[2], (_Float16)a1[3] };
        *(v8h*)&As[ar][ac0] = ah;

        // ---- B tile: 8 coalesced b32 loads (one column), transposed b128 store ----
        float bv0 = Bp[0 * (size_t)Nc];
        float bv1 = Bp[1 * (size_t)Nc];
        float bv2 = Bp[2 * (size_t)Nc];
        float bv3 = Bp[3 * (size_t)Nc];
        float bv4 = Bp[4 * (size_t)Nc];
        float bv5 = Bp[5 * (size_t)Nc];
        float bv6 = Bp[6 * (size_t)Nc];
        float bv7 = Bp[7 * (size_t)Nc];
        v8h bh = { (_Float16)bv0, (_Float16)bv1, (_Float16)bv2, (_Float16)bv3,
                   (_Float16)bv4, (_Float16)bv5, (_Float16)bv6, (_Float16)bv7 };
        *(v8h*)&Bt[bc][bk0] = bh;

        __syncthreads();

        // ---- fragments: all 16B ds_load_b128 ----
        const _Float16* arow = &As[wm * 16 + r16][0];
        v8h alo = *(const v8h*)(arow + half * 16);
        v8h ahi = *(const v8h*)(arow + half * 16 + 8);
        v16h a = __builtin_shufflevector(alo, ahi,
                    0,1,2,3,4,5,6,7,8,9,10,11,12,13,14,15);

        const _Float16* b0row = &Bt[wn * 32 + r16][half * 16];
        v8h b0lo = *(const v8h*)(b0row);
        v8h b0hi = *(const v8h*)(b0row + 8);
        v16h b0 = __builtin_shufflevector(b0lo, b0hi,
                    0,1,2,3,4,5,6,7,8,9,10,11,12,13,14,15);

        const _Float16* b1row = &Bt[wn * 32 + 16 + r16][half * 16];
        v8h b1lo = *(const v8h*)(b1row);
        v8h b1hi = *(const v8h*)(b1row + 8);
        v16h b1 = __builtin_shufflevector(b1lo, b1hi,
                    0,1,2,3,4,5,6,7,8,9,10,11,12,13,14,15);

        c0 = __builtin_amdgcn_wmma_f32_16x16x32_f16(false, a, false, b0,
                                                    (short)0, c0, false, false);
        c1 = __builtin_amdgcn_wmma_f32_16x16x32_f16(false, a, false, b1,
                                                    (short)0, c1, false, false);
        __syncthreads();

        Ap += 32;
        Bp += (size_t)32 * Nc;
    }

#pragma unroll
    for (int r = 0; r < 8; ++r) {
        int row = row0 + wm * 16 + r + 8 * half;
        if (row < M) {
            int colA = col0 + wn * 32 + r16;
            int colB = colA + 16;
            if (colA < Nc) {
                float v = c0[r] + (bias ? bias[colA] : 0.0f);
                if (act == 1) v = v > 0.f ? v : 0.f;
                C[(size_t)row * Nc + colA] = v;
            }
            if (colB < Nc) {
                float v = c1[r] + (bias ? bias[colB] : 0.0f);
                if (act == 1) v = v > 0.f ? v : 0.f;
                C[(size_t)row * Nc + colB] = v;
            }
        }
    }
}

// ---------------------------------------------------------------------
__global__ void zero_kernel(float* __restrict__ p, int n)
{
    int i = blockIdx.x * blockDim.x + threadIdx.x;
    if (i < n) p[i] = 0.0f;
}

// Per-(node,head) attention coefficients: one wave per (node,head).
__global__ __launch_bounds__(256)
void att_kernel(const float* __restrict__ hw, const float* __restrict__ att_s,
                const float* __restrict__ att_d, float* __restrict__ asrc,
                float* __restrict__ adst, int N, int H, int lgH, int outC)
{
    int w = blockIdx.x * 8 + (threadIdx.x >> 5);
    int lane = threadIdx.x & 31;
    if (w >= N * H) return;
    int node = w >> lgH, head = w & (H - 1);
    const float* hp = hw + (size_t)node * outC + head * HIDC;
    const float* as = att_s + head * HIDC;
    const float* ad = att_d + head * HIDC;
    float s1 = 0.f, s2 = 0.f;
#pragma unroll
    for (int i = 0; i < 4; ++i) {
        int c = lane + 32 * i;
        float v = hp[c];
        s1 += v * as[c];
        s2 += v * ad[c];
    }
    for (int o = 16; o > 0; o >>= 1) {
        s1 += __shfl_xor(s1, o);
        s2 += __shfl_xor(s2, o);
    }
    if (lane == 0) { asrc[w] = s1; adst[w] = s2; }
}

// Order-preserving float<->uint for hardware atomicMax.
__device__ inline unsigned f32_ord(float f)
{
    unsigned u = __float_as_uint(f);
    return (u & 0x80000000u) ? ~u : (u | 0x80000000u);
}
__device__ inline float ord_f32(unsigned u)
{
    return __uint_as_float((u & 0x80000000u) ? (u & 0x7fffffffu) : ~u);
}

__global__ void edge_max_kernel(const int* __restrict__ src, const int* __restrict__ dst,
                                const float* __restrict__ asrc, const float* __restrict__ adst,
                                unsigned* __restrict__ smax, int E, int N, int H, int lgH)
{
    int i = blockIdx.x * blockDim.x + threadIdx.x;
    int EE = E + N;
    if (i >= EE * H) return;
    int e = i >> lgH, h = i & (H - 1);
    int s, d;
    if (e < E) { s = src[e]; d = dst[e]; } else { s = e - E; d = s; }
    float el = asrc[s * H + h] + adst[d * H + h];
    el = el > 0.f ? el : 0.2f * el;               // leaky_relu(0.2)
    atomicMax(&smax[d * H + h], f32_ord(el));
}

__global__ void edge_exp_kernel(const int* __restrict__ src, const int* __restrict__ dst,
                                const float* __restrict__ asrc, const float* __restrict__ adst,
                                const unsigned* __restrict__ smax, float* __restrict__ ssum,
                                float* __restrict__ ew, int E, int N, int H, int lgH)
{
    int i = blockIdx.x * blockDim.x + threadIdx.x;
    int EE = E + N;
    if (i >= EE * H) return;
    int e = i >> lgH, h = i & (H - 1);
    int s, d;
    if (e < E) { s = src[e]; d = dst[e]; } else { s = e - E; d = s; }
    float el = asrc[s * H + h] + adst[d * H + h];
    el = el > 0.f ? el : 0.2f * el;
    float m = ord_f32(smax[d * H + h]);
    float w = __expf(el - m);
    ew[i] = w;
    atomicAdd(&ssum[d * H + h], w);
}

// One wave per (edge,head): 128-channel coalesced gather + atomic scatter.
__global__ __launch_bounds__(256)
void scatter_kernel(const int* __restrict__ src, const int* __restrict__ dst,
                    const float* __restrict__ hw, const float* __restrict__ ew,
                    const float* __restrict__ ssum, float* __restrict__ acc,
                    int E, int N, int H, int lgH, int outC)
{
    int w = blockIdx.x * 8 + (threadIdx.x >> 5);
    int lane = threadIdx.x & 31;
    int EE = E + N;
    if (w >= EE * H) return;
    int e = w >> lgH, h = w & (H - 1);
    int s, d;
    if (e < E) { s = src[e]; d = dst[e]; } else { s = e - E; d = s; }
    float alpha = ew[w] / (ssum[d * H + h] + 1e-16f);
    const float* hs = hw + (size_t)s * outC + h * HIDC;
    float* ad = acc + (size_t)d * outC + h * HIDC;
#pragma unroll
    for (int i = 0; i < 4; ++i) {
        int c = lane + 32 * i;
        atomicAdd(&ad[c], hs[c] * alpha);
    }
}

// BatchNorm statistics: per-block 64-row strip, LDS float atomics (ds_add_f32),
// then one global atomicAdd per channel per block.
__global__ __launch_bounds__(256)
void bn_stats_kernel(const float* __restrict__ x, float* __restrict__ gsum,
                     float* __restrict__ gsq, int N, int outC, int lg)
{
    __shared__ float sS[512];
    __shared__ float sQ[512];
    int tid = threadIdx.x;
    for (int c = tid; c < outC; c += 256) { sS[c] = 0.f; sQ[c] = 0.f; }
    __syncthreads();
    int row0 = blockIdx.x * 64;
    int mask = outC - 1;
    for (int idx = tid; idx < 64 * outC; idx += 256) {
        int r = idx >> lg, c = idx & mask;
        int row = row0 + r;
        if (row < N) {
            float v = x[(size_t)row * outC + c];
            atomicAdd(&sS[c], v);
            atomicAdd(&sQ[c], v * v);
        }
    }
    __syncthreads();
    for (int c = tid; c < outC; c += 256) {
        atomicAdd(&gsum[c], sS[c]);
        atomicAdd(&gsq[c], sQ[c]);
    }
}

__global__ void bn_finalize_kernel(const float* __restrict__ gsum, const float* __restrict__ gsq,
                                   const float* __restrict__ g, const float* __restrict__ b,
                                   float* __restrict__ scale, float* __restrict__ shift,
                                   int outC, int N)
{
    int c = blockIdx.x * blockDim.x + threadIdx.x;
    if (c >= outC) return;
    float inv = 1.0f / (float)N;
    float mean = gsum[c] * inv;
    float var = gsq[c] * inv - mean * mean;
    float sc = g[c] * rsqrtf(var + 1e-5f);
    scale[c] = sc;
    shift[c] = b[c] - mean * sc;
}

__global__ void bn_apply_kernel(const float* __restrict__ x, const float* __restrict__ scale,
                                const float* __restrict__ shift, float* __restrict__ y,
                                int n, int mask, int elu)
{
    int i = blockIdx.x * blockDim.x + threadIdx.x;
    if (i >= n) return;
    int c = i & mask;
    float v = x[i] * scale[c] + shift[c];
    if (elu) v = v > 0.f ? v : (__expf(v) - 1.0f);
    y[i] = v;
}

// risk = sigmoid(t1[N,64] @ Wr2[64,1] + br2): one wave per node.
__global__ __launch_bounds__(256)
void risk_kernel(const float* __restrict__ t1, const float* __restrict__ Wr2,
                 const float* __restrict__ br2, float* __restrict__ out, int N)
{
    int w = blockIdx.x * 8 + (threadIdx.x >> 5);
    int lane = threadIdx.x & 31;
    if (w >= N) return;
    const float* row = t1 + (size_t)w * 64;
    float s = row[lane] * Wr2[lane] + row[lane + 32] * Wr2[lane + 32];
    for (int o = 16; o > 0; o >>= 1) s += __shfl_xor(s, o);
    if (lane == 0) {
        float v = s + br2[0];
        out[w] = 1.0f / (1.0f + __expf(-v));
    }
}

// community = t2[N,64] @ Wc2[64,10] + bc2: thread per (node, j).
__global__ void comm_kernel(const float* __restrict__ t2, const float* __restrict__ Wc2,
                            const float* __restrict__ bc2, float* __restrict__ out, int N)
{
    int i = blockIdx.x * blockDim.x + threadIdx.x;
    if (i >= N * 10) return;
    int n = i / 10, j = i - n * 10;
    const float* row = t2 + (size_t)n * 64;
    float s = bc2[j];
#pragma unroll
    for (int k = 0; k < 64; ++k) s += row[k] * Wc2[k * 10 + j];
    out[i] = s;
}

// =====================================================================
extern "C" void kernel_launch(void* const* d_in, const int* in_sizes, int n_in,
                              void* d_out, int out_size, void* d_ws, size_t ws_size,
                              hipStream_t stream)
{
    (void)out_size; (void)ws_size;
    int N = 20000, E = 320000;
    if (n_in > 42) { N = in_sizes[42] / 64; E = in_sizes[1] / 2; }
    const int EE = E + N;

    // --- input unflattening (JAX pytree order: sorted dict keys) ---
    const int*   ei   = (const int*)d_in[1];
    const int*   srcI = ei;
    const int*   dstI = ei + E;
    const float* W_in = (const float*)d_in[2];
    const float* Wc1  = (const float*)d_in[3];
    const float* Wc2  = (const float*)d_in[4];
    const float* Wp   = (const float*)d_in[7];
    const float* Wr1  = (const float*)d_in[8];
    const float* Wr2  = (const float*)d_in[9];
    const float* b_in = (const float*)d_in[10];
    const float* bc1  = (const float*)d_in[11];
    const float* bc2  = (const float*)d_in[12];
    const float* bn_b[4] = {(const float*)d_in[15], (const float*)d_in[16],
                            (const float*)d_in[17], (const float*)d_in[18]};
    const float* bn_g[4] = {(const float*)d_in[19], (const float*)d_in[20],
                            (const float*)d_in[21], (const float*)d_in[22]};
    const float* bp   = (const float*)d_in[23];
    const float* br1  = (const float*)d_in[24];
    const float* br2  = (const float*)d_in[25];
    const float* gatW[4]  = {(const float*)d_in[26], (const float*)d_in[30],
                             (const float*)d_in[34], (const float*)d_in[38]};
    const float* att_d[4] = {(const float*)d_in[27], (const float*)d_in[31],
                             (const float*)d_in[35], (const float*)d_in[39]};
    const float* att_s[4] = {(const float*)d_in[28], (const float*)d_in[32],
                             (const float*)d_in[36], (const float*)d_in[40]};
    const float* x = (const float*)d_in[42];

    // --- workspace layout (floats) ---
    float* ws = (float*)d_ws;
    size_t o = 0;
    float* h    = ws + o; o += (size_t)N * 512;
    float* hw   = ws + o; o += (size_t)N * 512;
    float* acc  = ws + o; o += (size_t)N * 512;
    float* asrc = ws + o; o += (size_t)N * 4;
    float* adst = ws + o; o += (size_t)N * 4;
    unsigned* smax = (unsigned*)(ws + o); o += (size_t)N * 4;
    float* ssum = ws + o; o += (size_t)N * 4;
    float* ew   = ws + o; o += (size_t)EE * 4;
    float* bns  = ws + o; o += 512;           // bns/bnq contiguous (zeroed together)
    float* bnq  = ws + o; o += 512;
    float* scale = ws + o; o += 512;
    float* shift = ws + o; o += 512;

    auto cdiv = [](int a, int b) { return (a + b - 1) / b; };

    // input linear: h = relu(x @ W_in + b_in)
    gemm_wmma<<<dim3(cdiv(128, 64), cdiv(N, 64)), 256, 0, stream>>>(
        x, W_in, b_in, h, N, 64, 128, 1);

    int in_ch = 128;
    for (int L = 0; L < 4; ++L) {
        int H = (L < 3) ? 4 : 1;
        int lgH = (L < 3) ? 2 : 0;
        int outC = H * 128;

        // hW = h @ W  (GAT bias skipped: cancels exactly in batchnorm)
        gemm_wmma<<<dim3(cdiv(outC, 64), cdiv(N, 64)), 256, 0, stream>>>(
            h, gatW[L], nullptr, hw, N, in_ch, outC, 0);

        zero_kernel<<<cdiv(N * outC, 256), 256, 0, stream>>>(acc, N * outC);
        zero_kernel<<<cdiv(N * H, 256), 256, 0, stream>>>((float*)smax, N * H);
        zero_kernel<<<cdiv(N * H, 256), 256, 0, stream>>>(ssum, N * H);
        zero_kernel<<<4, 256, 0, stream>>>(bns, 1024);

        att_kernel<<<cdiv(N * H, 8), 256, 0, stream>>>(
            hw, att_s[L], att_d[L], asrc, adst, N, H, lgH, outC);
        edge_max_kernel<<<cdiv(EE * H, 256), 256, 0, stream>>>(
            srcI, dstI, asrc, adst, smax, E, N, H, lgH);
        edge_exp_kernel<<<cdiv(EE * H, 256), 256, 0, stream>>>(
            srcI, dstI, asrc, adst, smax, ssum, ew, E, N, H, lgH);
        scatter_kernel<<<cdiv(EE * H, 8), 256, 0, stream>>>(
            srcI, dstI, hw, ew, ssum, acc, E, N, H, lgH, outC);

        int lg = (outC == 512) ? 9 : 7;
        bn_stats_kernel<<<cdiv(N, 64), 256, 0, stream>>>(acc, bns, bnq, N, outC, lg);
        bn_finalize_kernel<<<cdiv(outC, 256), 256, 0, stream>>>(
            bns, bnq, bn_g[L], bn_b[L], scale, shift, outC, N);
        bn_apply_kernel<<<cdiv(N * outC, 256), 256, 0, stream>>>(
            acc, scale, shift, h, N * outC, outC - 1, (L < 3) ? 1 : 0);

        in_ch = outC;
    }

    // ---- output heads (final h is [N,128]) ----
    float* t1 = hw;                    // [N,64]
    float* t2 = hw + (size_t)N * 64;   // [N,64]
    float* emb  = (float*)d_out;                 // [N,32]
    float* risk = emb + (size_t)N * 32;          // [N]
    float* comm = risk + N;                      // [N,10]

    gemm_wmma<<<dim3(1, cdiv(N, 64)), 256, 0, stream>>>(h, Wr1, br1, t1, N, 128, 64, 1);
    risk_kernel<<<cdiv(N, 8), 256, 0, stream>>>(t1, Wr2, br2, risk, N);
    gemm_wmma<<<dim3(1, cdiv(N, 64)), 256, 0, stream>>>(h, Wc1, bc1, t2, N, 128, 64, 1);
    comm_kernel<<<cdiv(N * 10, 256), 256, 0, stream>>>(t2, Wc2, bc2, comm, N);
    gemm_wmma<<<dim3(1, cdiv(N, 64)), 256, 0, stream>>>(h, Wp, bp, emb, N, 128, 32, 0);
}
